// SelfAttention_47459388621442
// MI455X (gfx1250) — matrix operand order repaired and checked by reading, not compiled, not run on hardware
//
#include <hip/hip_runtime.h>
#include <hip/hip_bf16.h>
#include <stdint.h>

typedef __attribute__((ext_vector_type(16))) __bf16 v16bf;
typedef __attribute__((ext_vector_type(8)))  __bf16 v8bf;
typedef __attribute__((ext_vector_type(4)))  __bf16 v4bf;
typedef __attribute__((ext_vector_type(8)))  float  v8f;
typedef __attribute__((ext_vector_type(4)))  unsigned int v4u;
typedef __attribute__((ext_vector_type(8)))  int v8i;
typedef __attribute__((ext_vector_type(4)))  int v4i;

constexpr int CB  = 4;
constexpr int CS  = 2048;
constexpr int CD  = 1024;
constexpr int CH  = 16;
constexpr int CHD = 64;
constexpr int CM  = CB * CS;   // 8192 rows

// Assemble a 16-half fragment from two aligned 16-byte chunks (2x b128 loads).
__device__ __forceinline__ v16bf ld16(const __bf16* p0, const __bf16* p1) {
    v8bf lo = *(const v8bf*)p0;
    v8bf hi = *(const v8bf*)p1;
    return __builtin_shufflevector(lo, hi, 0,1,2,3,4,5,6,7,8,9,10,11,12,13,14,15);
}

// ---------------------------------------------------------------------------
// QKV projection: dst[b,h,s,e] = (X[m,:] @ W[:,n] + bias[n]) as bf16.
// Block tile 128x128xK32, 8 waves; each wave owns a 32x64 tile (8 WMMA accums).
// ---------------------------------------------------------------------------
__global__ __launch_bounds__(256) void qkv_proj_kernel(
    const float* __restrict__ X, const float* __restrict__ W,
    const float* __restrict__ bias, __bf16* __restrict__ dst)
{
    __shared__ __align__(16) __bf16 As [128 * 40];  // [m][k], 80B rows
    __shared__ __align__(16) __bf16 Bst[128 * 40];  // [n][k] transposed

    const uint32_t t    = threadIdx.x;
    const uint32_t lane = t & 31;
    const uint32_t w    = t >> 5;
    const uint32_t wrow = w >> 1;       // 0..3 -> rows wrow*32
    const uint32_t wcol = w & 1;        // 0..1 -> cols wcol*64
    const uint32_t gm0  = blockIdx.y * 128;
    const uint32_t gn0  = blockIdx.x * 128;

    const uint32_t mA   = lane & 15;
    const uint32_t kbA  = (lane < 16) ? 0 : 8;
    const uint32_t nB   = lane & 15;
    const uint32_t kbB  = (lane < 16) ? 0 : 8;

    const v8f vzero = {};
    v8f acc[2][4];
    #pragma unroll
    for (int rt = 0; rt < 2; ++rt)
        #pragma unroll
        for (int ct = 0; ct < 4; ++ct) acc[rt][ct] = vzero;

    for (uint32_t k0 = 0; k0 < CD; k0 += 32) {
        // stage A 128x32 (f32->bf16), 4 float4 chunks per thread
        #pragma unroll
        for (int j = 0; j < 4; ++j) {
            uint32_t chunk = t + j * 256;          // 0..1023
            uint32_t r = chunk >> 3, c4 = (chunk & 7) * 4;
            float4 x = *(const float4*)&X[(size_t)(gm0 + r) * CD + k0 + c4];
            v4bf p = { (__bf16)x.x, (__bf16)x.y, (__bf16)x.z, (__bf16)x.w };
            *(v4bf*)&As[r * 40 + c4] = p;
        }
        // stage B 32x128 transposed -> Bst[n][k]
        #pragma unroll
        for (int j = 0; j < 4; ++j) {
            uint32_t chunk = t + j * 256;
            uint32_t kk = chunk >> 5, n4 = (chunk & 31) * 4;
            float4 x = *(const float4*)&W[(size_t)(k0 + kk) * CD + gn0 + n4];
            Bst[(n4 + 0) * 40 + kk] = (__bf16)x.x;
            Bst[(n4 + 1) * 40 + kk] = (__bf16)x.y;
            Bst[(n4 + 2) * 40 + kk] = (__bf16)x.z;
            Bst[(n4 + 3) * 40 + kk] = (__bf16)x.w;
        }
        if (k0 + 32 < CD) {
            __builtin_prefetch(&X[(size_t)(gm0 + (t & 127)) * CD + k0 + 32 + (t >> 7) * 16], 0, 3);
            __builtin_prefetch(&W[(size_t)(k0 + 32 + (t & 31)) * CD + gn0 + (t >> 5) * 16], 0, 3);
        }
        __syncthreads();

        const __bf16* ar0 = &As[(wrow * 32 + mA) * 40];
        const __bf16* ar1 = &As[(wrow * 32 + 16 + mA) * 40];
        v16bf a0 = ld16(ar0 + kbA, ar0 + 16 + kbA);
        v16bf a1 = ld16(ar1 + kbA, ar1 + 16 + kbA);
        #pragma unroll
        for (uint32_t ct = 0; ct < 4; ++ct) {
            const __bf16* br = &Bst[(wcol * 64 + ct * 16 + nB) * 40 + kbB];
            v16bf bfr = ld16(br, br + 16);
            acc[0][ct] = __builtin_amdgcn_wmma_f32_16x16x32_bf16(false, a0, false, bfr, (short)0, acc[0][ct], false, false);
            acc[1][ct] = __builtin_amdgcn_wmma_f32_16x16x32_bf16(false, a1, false, bfr, (short)0, acc[1][ct], false, false);
        }
        __syncthreads();
    }

    const uint32_t roff = (lane < 16) ? 0 : 8;
    #pragma unroll
    for (uint32_t rt = 0; rt < 2; ++rt) {
        #pragma unroll
        for (uint32_t r = 0; r < 8; ++r) {
            uint32_t grow = gm0 + wrow * 32 + rt * 16 + roff + r;
            uint32_t bb = grow >> 11, ss = grow & 2047;        // /CS, %CS
            #pragma unroll
            for (uint32_t ct = 0; ct < 4; ++ct) {
                uint32_t gcol = gn0 + wcol * 64 + ct * 16 + nB;
                float val = acc[rt][ct][r] + bias[gcol];
                uint32_t hh = gcol >> 6, ee = gcol & 63;       // /HD, %HD
                dst[(((size_t)bb * CH + hh) * CS + ss) * CHD + ee] = (__bf16)val;
            }
        }
    }
}

// ---------------------------------------------------------------------------
// Flash attention per (b,h): 256 Q rows/block (32 per wave).  K tile staged
// by the Tensor Data Mover (TDM) into LDS (row-major + pad), V transposed
// manually, online softmax, output to [B,S,H,HD] bf16.
// ---------------------------------------------------------------------------
__global__ __launch_bounds__(256) void flash_attn_kernel(
    const __bf16* __restrict__ Qh, const __bf16* __restrict__ Kh,
    const __bf16* __restrict__ Vh, __bf16* __restrict__ Oh)
{
    __shared__ __align__(16) __bf16 Ks[64 * 72];        // [key][hd], 144B rows
    __shared__ __align__(16) __bf16 Vt[64 * 72];        // [hd][key] transposed
    __shared__ __align__(16) __bf16 Ps[8 * 32 * 64];    // per-wave P scratch

    const uint32_t t    = threadIdx.x;
    const uint32_t lane = t & 31;
    const uint32_t w    = t >> 5;
    const uint32_t bh   = blockIdx.y;
    const uint32_t b    = bh >> 4;     // /CH
    const uint32_t h    = bh & 15;     // %CH
    const size_t head_off = (size_t)bh * CS * CHD;
    const __bf16* Qp = Qh + head_off;
    const __bf16* Kp = Kh + head_off;
    const __bf16* Vp = Vh + head_off;

    const uint32_t qrow0 = blockIdx.x * 256 + w * 32;
    const uint32_t mA   = lane & 15;
    const uint32_t kbA  = (lane < 16) ? 0 : 8;
    const uint32_t nB   = lane & 15;
    const uint32_t kbB  = (lane < 16) ? 0 : 8;
    const uint32_t roff = (lane < 16) ? 0 : 8;

    // TDM descriptor group1: bf16 (data_size=1 -> 2B), 2D tensor 64-wide rows,
    // tensor_dim0=64, dim0_stride=64, tensor_dim1=2048, tile 64x64,
    // LDS pad: every 32 DWORDs (128B row) insert 4 DWORDs (-> 72-half rows).
    v8i tg1;
    tg1[0] = (1 << 16) | (1 << 20) | (4 << 22) | (3 << 25);
    tg1[1] = (int)(64u << 16);     // tensor_dim0[15:0] at [31:16]
    tg1[2] = (int)(2048u << 16);   // dim0[31:16]=0 | tensor_dim1[15:0]
    tg1[3] = (int)(64u << 16);     // tensor_dim1[31:16]=0 | tile_dim0=64
    tg1[4] = 64;                   // tile_dim1=64 | tile_dim2=0
    tg1[5] = 64;                   // tensor_dim0_stride[31:0]=64
    tg1[6] = 0;
    tg1[7] = 0;
    v4i tzero4;
    tzero4[0] = 0; tzero4[1] = 0; tzero4[2] = 0; tzero4[3] = 0;
    v8i tzero8;
    #pragma unroll
    for (int i = 0; i < 8; ++i) tzero8[i] = 0;
    const uint32_t ks_lds =
        (uint32_t)(uintptr_t)(__attribute__((address_space(3))) __bf16*)&Ks[0];

    // Q fragments: 2 row-tiles x 2 hd-chunks, loaded once from global.
    v16bf qa[2][2];
    #pragma unroll
    for (uint32_t rt = 0; rt < 2; ++rt) {
        const __bf16* qrow = Qp + (size_t)(qrow0 + rt * 16 + mA) * CHD;
        qa[rt][0] = ld16(qrow + kbA,      qrow + 16 + kbA);
        qa[rt][1] = ld16(qrow + 32 + kbA, qrow + 48 + kbA);
    }

    const v8f vzero = {};
    float mrow[2][8], lrow[2][8];
    v8f oacc[2][4];
    #pragma unroll
    for (int rt = 0; rt < 2; ++rt) {
        #pragma unroll
        for (int r = 0; r < 8; ++r) { mrow[rt][r] = -3.0e38f; lrow[rt][r] = 0.0f; }
        #pragma unroll
        for (int c = 0; c < 4; ++c) oacc[rt][c] = vzero;
    }

    const float scale = 0.125f;   // 1/sqrt(64)

    for (uint32_t kt0 = 0; kt0 < CS; kt0 += 64) {
        // K tile via Tensor Data Mover (one wave issues the DMA)
        if (w == 0) {
            uint64_t ga = (uint64_t)(uintptr_t)(Kp + (size_t)kt0 * CHD);
            v4u tg0;
            tg0[0] = 1u;                                   // count=1
            tg0[1] = ks_lds;                               // lds_addr
            tg0[2] = (unsigned)ga;                         // global_addr[31:0]
            tg0[3] = ((unsigned)(ga >> 32) & 0x01FFFFFFu)  // global_addr[56:32]
                     | (2u << 30);                         // type=2 (image)
            __builtin_amdgcn_tensor_load_to_lds(tg0, tg1, tzero4, tzero4, tzero8, 0);
        }
        // stage V (transposed) 64x64 tile manually
        #pragma unroll
        for (int j = 0; j < 2; ++j) {
            uint32_t idx = t + j * 256;            // 0..511 chunks of 8 halfs
            uint32_t r = idx >> 3;                 // key
            uint32_t c = (idx & 7) * 8;            // hd offset
            v8bf x = *(const v8bf*)&Vp[(size_t)(kt0 + r) * CHD + c];
            #pragma unroll
            for (int i = 0; i < 8; ++i) Vt[(c + i) * 72 + r] = x[i];
        }
        if (kt0 + 64 < CS)
            __builtin_prefetch(Vp + (size_t)(kt0 + 64) * CHD + t * 32, 0, 3);
        if (w == 0) __builtin_amdgcn_s_wait_tensorcnt(0);
        __syncthreads();

        // S = Q @ K^T : B-fragments shared across both Q row-tiles
        v8f sacc[2][4];
        #pragma unroll
        for (int rt = 0; rt < 2; ++rt)
            #pragma unroll
            for (int c = 0; c < 4; ++c) sacc[rt][c] = vzero;
        #pragma unroll
        for (uint32_t c = 0; c < 4; ++c) {
            const __bf16* krow = &Ks[(c * 16 + nB) * 72];
            v16bf kb0 = ld16(krow + kbB,      krow + kbB + 16);
            v16bf kb1 = ld16(krow + 32 + kbB, krow + 48 + kbB);
            #pragma unroll
            for (uint32_t rt = 0; rt < 2; ++rt) {
                sacc[rt][c] = __builtin_amdgcn_wmma_f32_16x16x32_bf16(false, qa[rt][0], false, kb0, (short)0, sacc[rt][c], false, false);
                sacc[rt][c] = __builtin_amdgcn_wmma_f32_16x16x32_bf16(false, qa[rt][1], false, kb1, (short)0, sacc[rt][c], false, false);
            }
        }

        // online softmax per row-tile
        #pragma unroll
        for (uint32_t rt = 0; rt < 2; ++rt) {
            #pragma unroll
            for (int c = 0; c < 4; ++c)
                #pragma unroll
                for (int r = 0; r < 8; ++r) sacc[rt][c][r] *= scale;

            float mnew[8], corr[8], psum[8];
            #pragma unroll
            for (int r = 0; r < 8; ++r) {
                float tm = fmaxf(fmaxf(sacc[rt][0][r], sacc[rt][1][r]),
                                 fmaxf(sacc[rt][2][r], sacc[rt][3][r]));
                #pragma unroll
                for (int msk = 1; msk < 16; msk <<= 1)
                    tm = fmaxf(tm, __shfl_xor(tm, msk, 32));
                mnew[r] = fmaxf(mrow[rt][r], tm);
                corr[r] = __expf(mrow[rt][r] - mnew[r]);
                mrow[rt][r] = mnew[r];
                psum[r] = 0.0f;
            }
            #pragma unroll
            for (uint32_t c = 0; c < 4; ++c) {
                #pragma unroll
                for (uint32_t r = 0; r < 8; ++r) {
                    float p = __expf(sacc[rt][c][r] - mnew[r]);
                    psum[r] += p;
                    Ps[w * 2048 + (rt * 16 + roff + r) * 64 + c * 16 + nB] = (__bf16)p;
                }
            }
            #pragma unroll
            for (int r = 0; r < 8; ++r) {
                float s = psum[r];
                #pragma unroll
                for (int msk = 1; msk < 16; msk <<= 1)
                    s += __shfl_xor(s, msk, 32);
                lrow[rt][r] = lrow[rt][r] * corr[r] + s;
                #pragma unroll
                for (int c = 0; c < 4; ++c) oacc[rt][c][r] *= corr[r];
            }
        }

        // O += P @ V   (vb fragments shared across both row-tiles)
        #pragma unroll
        for (uint32_t kc = 0; kc < 2; ++kc) {
            v16bf pa[2];
            #pragma unroll
            for (uint32_t rt = 0; rt < 2; ++rt) {
                const __bf16* prow = &Ps[w * 2048 + (rt * 16 + mA) * 64 + kc * 32];
                pa[rt] = ld16(prow + kbA, prow + 16 + kbA);
            }
            #pragma unroll
            for (uint32_t c = 0; c < 4; ++c) {
                const __bf16* vrow = &Vt[(c * 16 + nB) * 72 + kc * 32 + kbB];
                v16bf vb = ld16(vrow, vrow + 16);
                #pragma unroll
                for (uint32_t rt = 0; rt < 2; ++rt)
                    oacc[rt][c] = __builtin_amdgcn_wmma_f32_16x16x32_bf16(false, pa[rt], false, vb, (short)0, oacc[rt][c], false, false);
            }
        }
        __syncthreads();
    }

    #pragma unroll
    for (uint32_t rt = 0; rt < 2; ++rt) {
        #pragma unroll
        for (uint32_t r = 0; r < 8; ++r) {
            float inv = 1.0f / lrow[rt][r];
            uint32_t srow = qrow0 + rt * 16 + roff + r;
            #pragma unroll
            for (uint32_t c = 0; c < 4; ++c) {
                size_t oidx = (((size_t)b * CS + srow) * CH + h) * CHD + c * 16 + nB;
                Oh[oidx] = (__bf16)(oacc[rt][c][r] * inv);
            }
        }
    }
}

// ---------------------------------------------------------------------------
// Output projection: d_out[m,n] = Xbf[m,:] @ Wo[:,n] + bo[n]  (f32 out)
// Same 128x128 block tile / 32x64 wave tile as the QKV GEMM.
// ---------------------------------------------------------------------------
__global__ __launch_bounds__(256) void out_proj_kernel(
    const __bf16* __restrict__ Xbf, const float* __restrict__ W,
    const float* __restrict__ bias, float* __restrict__ dst)
{
    __shared__ __align__(16) __bf16 As [128 * 40];
    __shared__ __align__(16) __bf16 Bst[128 * 40];   // [n][k] transposed

    const uint32_t t    = threadIdx.x;
    const uint32_t lane = t & 31;
    const uint32_t w    = t >> 5;
    const uint32_t wrow = w >> 1;
    const uint32_t wcol = w & 1;
    const uint32_t gm0  = blockIdx.y * 128;
    const uint32_t gn0  = blockIdx.x * 128;

    const uint32_t mA   = lane & 15;
    const uint32_t kbA  = (lane < 16) ? 0 : 8;
    const uint32_t nB   = lane & 15;
    const uint32_t kbB  = (lane < 16) ? 0 : 8;

    const v8f vzero = {};
    v8f acc[2][4];
    #pragma unroll
    for (int rt = 0; rt < 2; ++rt)
        #pragma unroll
        for (int ct = 0; ct < 4; ++ct) acc[rt][ct] = vzero;

    for (uint32_t k0 = 0; k0 < CD; k0 += 32) {
        #pragma unroll
        for (int j = 0; j < 2; ++j) {
            uint32_t chunk = t + j * 256;          // 0..511
            uint32_t r = chunk >> 2, c8 = (chunk & 3) * 8;
            *(v8bf*)&As[r * 40 + c8] =
                *(const v8bf*)&Xbf[(size_t)(gm0 + r) * CD + k0 + c8];
        }
        #pragma unroll
        for (int j = 0; j < 4; ++j) {
            uint32_t chunk = t + j * 256;
            uint32_t kk = chunk >> 5, n4 = (chunk & 31) * 4;
            float4 x = *(const float4*)&W[(size_t)(k0 + kk) * CD + gn0 + n4];
            Bst[(n4 + 0) * 40 + kk] = (__bf16)x.x;
            Bst[(n4 + 1) * 40 + kk] = (__bf16)x.y;
            Bst[(n4 + 2) * 40 + kk] = (__bf16)x.z;
            Bst[(n4 + 3) * 40 + kk] = (__bf16)x.w;
        }
        if (k0 + 32 < CD) {
            __builtin_prefetch(&Xbf[(size_t)(gm0 + (t & 127)) * CD + k0 + 32], 0, 3);
            __builtin_prefetch(&W[(size_t)(k0 + 32 + (t & 31)) * CD + gn0 + (t >> 5) * 16], 0, 3);
        }
        __syncthreads();

        const __bf16* ar0 = &As[(wrow * 32 + mA) * 40];
        const __bf16* ar1 = &As[(wrow * 32 + 16 + mA) * 40];
        v16bf a0 = ld16(ar0 + kbA, ar0 + 16 + kbA);
        v16bf a1 = ld16(ar1 + kbA, ar1 + 16 + kbA);
        #pragma unroll
        for (uint32_t ct = 0; ct < 4; ++ct) {
            const __bf16* br = &Bst[(wcol * 64 + ct * 16 + nB) * 40 + kbB];
            v16bf bfr = ld16(br, br + 16);
            acc[0][ct] = __builtin_amdgcn_wmma_f32_16x16x32_bf16(false, a0, false, bfr, (short)0, acc[0][ct], false, false);
            acc[1][ct] = __builtin_amdgcn_wmma_f32_16x16x32_bf16(false, a1, false, bfr, (short)0, acc[1][ct], false, false);
        }
        __syncthreads();
    }

    const uint32_t roff = (lane < 16) ? 0 : 8;
    #pragma unroll
    for (uint32_t rt = 0; rt < 2; ++rt) {
        #pragma unroll
        for (uint32_t r = 0; r < 8; ++r) {
            uint32_t grow = gm0 + wrow * 32 + rt * 16 + roff + r;
            #pragma unroll
            for (uint32_t ct = 0; ct < 4; ++ct) {
                uint32_t gcol = gn0 + wcol * 64 + ct * 16 + nB;
                dst[(size_t)grow * CD + gcol] = acc[rt][ct][r] + bias[gcol];
            }
        }
    }
}

// ---------------------------------------------------------------------------
extern "C" void kernel_launch(void* const* d_in, const int* in_sizes, int n_in,
                              void* d_out, int out_size, void* d_ws, size_t ws_size,
                              hipStream_t stream)
{
    (void)in_sizes; (void)n_in; (void)out_size; (void)ws_size;
    const float* q  = (const float*)d_in[0];
    const float* k  = (const float*)d_in[1];
    const float* v  = (const float*)d_in[2];
    const float* Wq = (const float*)d_in[3];
    const float* bq = (const float*)d_in[4];
    const float* Wk = (const float*)d_in[5];
    const float* bk = (const float*)d_in[6];
    const float* Wv = (const float*)d_in[7];
    const float* bv = (const float*)d_in[8];
    const float* Wo = (const float*)d_in[9];
    const float* bo = (const float*)d_in[10];

    const size_t headElems = (size_t)CB * CH * CS * CHD;   // 8,388,608
    __bf16* qh = (__bf16*)d_ws;
    __bf16* kh = qh + headElems;
    __bf16* vh = kh + headElems;
    __bf16* oh = vh + headElems;

    dim3 gGemm(CD / 128, CM / 128);   // 8 x 64 blocks
    qkv_proj_kernel<<<gGemm, 256, 0, stream>>>(q, Wq, bq, qh);
    qkv_proj_kernel<<<gGemm, 256, 0, stream>>>(k, Wk, bk, kh);
    qkv_proj_kernel<<<gGemm, 256, 0, stream>>>(v, Wv, bv, vh);

    dim3 gAttn(CS / 256, CB * CH);    // 8 x 64 blocks
    flash_attn_kernel<<<gAttn, 256, 0, stream>>>(qh, kh, vh, oh);

    out_proj_kernel<<<gGemm, 256, 0, stream>>>(oh, Wo, bo, (float*)d_out);
}
